// VariationalGIB_4818953306903
// MI455X (gfx1250) — matrix operand order repaired and checked by reading, not compiled, not run on hardware
//
#include <hip/hip_runtime.h>
#include <hip/hip_bf16.h>
#include <math.h>

typedef __attribute__((ext_vector_type(16))) _Float16 v16h;
typedef __attribute__((ext_vector_type(8)))  _Float16 v8h;
typedef __attribute__((ext_vector_type(8)))  float    v8f;

// ---------------- problem constants ----------------
constexpr int G_   = 2048;
constexpr int P_   = 50;
constexpr int N_   = G_ * P_;        // 102400
constexpr int DEG_ = 8;
constexpr int EPG_ = DEG_ * P_;      // 400 edges per graph
constexpr int E_   = N_ * DEG_;      // 819200
constexpr int D_   = 128;            // F_IN = D1 = D2 = 128
constexpr int H1_  = 16;
constexpr int TILES_ = 4;            // row-tiles (of 16) per GEMM block
constexpr float BN_EPS_ = 1e-5f;
constexpr float EPS_    = 1e-7f;

// ---------------- utility kernels ----------------
__global__ void zero_kernel(float* stats, float* outsc) {
  int i = blockIdx.x * blockDim.x + threadIdx.x;
  if (i < 512) stats[i] = 0.f;       // sum1,sq1,sum2,sq2
  if (i < 3)   outsc[i] = 0.f;       // pos_penalty, kl, preserve
}

// Pack W (128x128 f32, row-major, fan_in x fan_out) into WMMA-B-fragment
// order: idx = (((kb*8 + wavetile)*2 + half)*16 + ncol_lo)*16 + j
// so each lane's 16 halfs are one contiguous, 32B-aligned chunk and
// consecutive lanes read consecutive chunks (coalesced b128 loads).
__global__ void pack_weight_kernel(const float* __restrict__ W,
                                   _Float16* __restrict__ dst) {
  int i = blockIdx.x * blockDim.x + threadIdx.x;   // 0 .. 16383
  if (i >= D_ * D_) return;
  int j    = i & 15;
  int nl   = (i >> 4) & 15;
  int half = (i >> 8) & 1;
  int wt   = (i >> 9) & 7;
  int kb   = (i >> 12) & 3;
  int k = kb * 32 + 8 * half + (j & 7) + 16 * (j >> 3);   // ISA B-layout: element j <-> K=8*half+{0..7,16..23}
  int n = wt * 16 + nl;
  dst[i] = (_Float16)W[k * D_ + n];
}

// ---------------- per-graph GIN aggregation (LDS) ----------------
// z[node] = h[node] + sum_{e: dst==node} h[src(e)]
// Optionally applies BN scale/shift + relu while staging h into LDS (layer 2).
__global__ __launch_bounds__(256) void gin_aggr_kernel(
    const float* __restrict__ hin,
    const int* __restrict__ esrc, const int* __restrict__ edst,
    const float* __restrict__ scale, const float* __restrict__ shift,
    int apply_bn_relu,
    float* __restrict__ zout) {
  __shared__ float hs[P_ * D_];   // 25.6 KB
  __shared__ float ag[P_ * D_];   // 25.6 KB
  __shared__ int   ls[EPG_];
  __shared__ int   ld[EPG_];
  const int g = blockIdx.x, tid = threadIdx.x;
  const int nodeBase = g * P_;
  for (int i = tid; i < P_ * D_; i += 256) {
    float v = hin[(size_t)nodeBase * D_ + i];
    if (apply_bn_relu) {
      int d = i & (D_ - 1);
      v = scale[d] * v + shift[d];
      v = v > 0.f ? v : 0.f;
    }
    hs[i] = v;
    ag[i] = 0.f;
  }
  const int eBase = g * EPG_;
  for (int i = tid; i < EPG_; i += 256) {
    ls[i] = esrc[eBase + i] - nodeBase;
    ld[i] = edst[eBase + i] - nodeBase;
  }
  __syncthreads();
  for (int i = tid; i < EPG_ * D_; i += 256) {
    int e = i >> 7, d = i & (D_ - 1);
    atomicAdd(&ag[ld[e] * D_ + d], hs[ls[e] * D_ + d]);   // ds_add_f32
  }
  __syncthreads();
  for (int i = tid; i < P_ * D_; i += 256)
    zout[(size_t)nodeBase * D_ + i] = hs[i] + ag[i];
}

// ---------------- fused 2-GEMM MLP via WMMA + BN-stat accumulation ----------
// u = relu(relu(z*Wa + ba)*Wb + bb); also accumulates column sum / sumsq.
// 256 threads = 8 waves; wave w owns output columns [16w, 16w+16).
// Each block processes TILES_ consecutive 16-row tiles, holding all packed
// B fragments of both weight matrices in registers across the tile loop.
__global__ __launch_bounds__(256) void mlp2_wmma_kernel(
    const float* __restrict__ zin,
    const _Float16* __restrict__ Wa, const float* __restrict__ ba,
    const _Float16* __restrict__ Wb, const float* __restrict__ bb,
    float* __restrict__ uout,
    float* __restrict__ statSum, float* __restrict__ statSq) {
  __shared__ __align__(16) _Float16 tA[16 * D_];   // input tile, f16
  __shared__ __align__(16) _Float16 tB[16 * D_];   // mid tile, f16
  const int tid  = threadIdx.x;
  const int lane = tid & 31;
  const int wv   = tid >> 5;
  const int half = lane >> 4;
  const int mrow = lane & 15;              // A row this lane covers
  const int ncol = 16 * wv + (lane & 15);  // output column this lane covers

  // ---- preload packed B fragments for both GEMMs (register-resident) ----
  v16h wa[4], wb[4];
#pragma unroll
  for (int k4 = 0; k4 < 4; ++k4) {
    size_t off = ((((size_t)k4 * 8 + wv) * 2 + half) * 16 + (lane & 15)) * 16;
    wa[k4] = *(const v16h*)(Wa + off);
    wb[k4] = *(const v16h*)(Wb + off);
  }
  const float bias1 = ba[ncol];
  const float bias2 = bb[ncol];

  float s = 0.f, ss = 0.f;
  for (int t = 0; t < TILES_; ++t) {
    const int    row0     = (blockIdx.x * TILES_ + t) * 16;
    const float* tileBase = zin + (size_t)row0 * D_;   // contiguous 16x128 block
    if (t + 1 < TILES_)
      __builtin_prefetch(tileBase + 16 * D_ + tid * 8, 0, 1); // global_prefetch
    __syncthreads();  // previous iteration's LDS readers are done

    // stage tile: 2048 f32 -> f16; thread moves one 32B chunk
    {
      const float4* zp = (const float4*)(tileBase + tid * 8);
      float4 f0 = zp[0], f1 = zp[1];
      v8h h;
      h[0] = (_Float16)f0.x; h[1] = (_Float16)f0.y;
      h[2] = (_Float16)f0.z; h[3] = (_Float16)f0.w;
      h[4] = (_Float16)f1.x; h[5] = (_Float16)f1.y;
      h[6] = (_Float16)f1.z; h[7] = (_Float16)f1.w;
      *(v8h*)(tA + tid * 8) = h;
    }
    __syncthreads();

    // ---- GEMM 1: tA (16x128) * Wa, K in 4 steps of 32 ----
    v8f c = {};
#pragma unroll
    for (int k4 = 0; k4 < 4; ++k4) {
      // ISA A-layout: lane holds K = 32*k4 + 8*half + {0..7, 16..23}
      v8h lo = *(const v8h*)(tA + mrow * D_ + 32 * k4 + 8 * half);
      v8h hi = *(const v8h*)(tA + mrow * D_ + 32 * k4 + 8 * half + 16);
      v16h a;
#pragma unroll
      for (int j = 0; j < 8; ++j) { a[j] = lo[j]; a[j + 8] = hi[j]; }
      c = __builtin_amdgcn_wmma_f32_16x16x32_f16(false, a, false, wa[k4],
                                                 (short)0, c, false, false);
    }
#pragma unroll
    for (int r = 0; r < 8; ++r) {
      float v = c[r] + bias1;
      v = v > 0.f ? v : 0.f;                          // relu
      tB[(r + 8 * half) * D_ + ncol] = (_Float16)v;   // M = r + 8*half
    }
    __syncthreads();

    // ---- GEMM 2: tB (16x128) * Wb ----
    v8f c2 = {};
#pragma unroll
    for (int k4 = 0; k4 < 4; ++k4) {
      v8h lo = *(const v8h*)(tB + mrow * D_ + 32 * k4 + 8 * half);
      v8h hi = *(const v8h*)(tB + mrow * D_ + 32 * k4 + 8 * half + 16);
      v16h a;
#pragma unroll
      for (int j = 0; j < 8; ++j) { a[j] = lo[j]; a[j + 8] = hi[j]; }
      c2 = __builtin_amdgcn_wmma_f32_16x16x32_f16(false, a, false, wb[k4],
                                                  (short)0, c2, false, false);
    }
#pragma unroll
    for (int r = 0; r < 8; ++r) {
      float v = c2[r] + bias2;
      v = v > 0.f ? v : 0.f;                          // relu
      uout[(size_t)(row0 + r + 8 * half) * D_ + ncol] = v;
      s += v;
      ss += v * v;
    }
  }
  // lanes l and l^16 hold the same column: pair-reduce then one atomic each
  s  += __shfl_xor(s, 16);
  ss += __shfl_xor(ss, 16);
  if (half == 0) {
    atomicAdd(&statSum[ncol], s);
    atomicAdd(&statSq[ncol], ss);
  }
}

// ---------------- BN parameter computation ----------------
__global__ void bn_params_kernel(const float* __restrict__ sum,
                                 const float* __restrict__ sq,
                                 const float* __restrict__ gamma,
                                 const float* __restrict__ beta,
                                 float* __restrict__ scale,
                                 float* __restrict__ shift) {
  int d = threadIdx.x;  // 128 threads
  float m   = sum[d] / (float)N_;
  float var = sq[d] / (float)N_ - m * m;   // biased var, matches h.var(0)
  float inv = rsqrtf(var + BN_EPS_);
  float sc  = gamma[d] * inv;
  scale[d] = sc;
  shift[d] = beta[d] - m * sc;
}

__global__ void bn_apply_kernel(const float* __restrict__ u,
                                const float* __restrict__ scale,
                                const float* __restrict__ shift,
                                float* __restrict__ nf) {
  size_t i = (size_t)blockIdx.x * blockDim.x + threadIdx.x;
  if (i < (size_t)N_ * D_) {
    int d = (int)(i & (D_ - 1));
    nf[i] = scale[d] * u[i] + shift[d];   // no relu for layer-2 output
  }
}

// ---------------- assignment head (tanh MLP + softmax + gumbel-softmax) ----
__global__ void head_kernel(const float* __restrict__ nf,
                            const float* __restrict__ Wf1,
                            const float* __restrict__ bf1,
                            const float* __restrict__ Wf2,
                            const float* __restrict__ bf2,
                            const float* __restrict__ gumbel,
                            float* __restrict__ assign,
                            float* __restrict__ lamb) {
  int n = blockIdx.x * blockDim.x + threadIdx.x;
  if (n >= N_) return;
  const float* row = nf + (size_t)n * D_;
  float ab[H1_];
#pragma unroll
  for (int o = 0; o < H1_; ++o) {
    float acc = bf1[o];
    for (int k = 0; k < D_; ++k) acc += row[k] * Wf1[k * H1_ + o];
    ab[o] = tanhf(acc);
  }
  float l0 = bf2[0], l1 = bf2[1];
#pragma unroll
  for (int o = 0; o < H1_; ++o) {
    l0 += ab[o] * Wf2[o * 2 + 0];
    l1 += ab[o] * Wf2[o * 2 + 1];
  }
  float mx = fmaxf(l0, l1);
  float e0 = __expf(l0 - mx), e1 = __expf(l1 - mx);
  float inv = 1.f / (e0 + e1);
  float a0 = e0 * inv, a1 = e1 * inv;
  assign[n * 2 + 0] = a0;
  assign[n * 2 + 1] = a1;
  float g0 = a0 + gumbel[n * 2 + 0], g1 = a1 + gumbel[n * 2 + 1];
  mx = fmaxf(g0, g1);
  e0 = __expf(g0 - mx); e1 = __expf(g1 - mx);
  inv = 1.f / (e0 + e1);
  lamb[n * 2 + 0] = e0 * inv;
  lamb[n * 2 + 1] = e1 * inv;
}

// ---------------- per-graph mean / std of nf ----------------
__global__ void gstats_kernel(const float* __restrict__ nf,
                              float* __restrict__ gmean,
                              float* __restrict__ gstd) {
  int g = blockIdx.x, d = threadIdx.x;   // 128 threads
  const float* base = nf + (size_t)g * P_ * D_ + d;
  float s = 0.f;
  for (int p = 0; p < P_; ++p) s += base[p * D_];
  float m = s / (float)P_;
  float q = 0.f;
  for (int p = 0; p < P_; ++p) { float v = base[p * D_] - m; q += v * v; }
  gmean[g * D_ + d] = m;
  gstd[g * D_ + d]  = sqrtf(q / (float)(P_ - 1));
}

// ---------------- per-graph embeddings + KL + preserve ----------------
__global__ __launch_bounds__(128) void final_kernel(
    const float* __restrict__ nf, const float* __restrict__ unif,
    const float* __restrict__ assign, const float* __restrict__ lamb,
    const float* __restrict__ gmean, const float* __restrict__ gstd,
    float* __restrict__ out) {
  __shared__ float redA[D_];
  __shared__ float redB[D_];
  const int g = blockIdx.x, d = threadIdx.x;
  const float m = gmean[g * D_ + d];
  const float s = gstd[g * D_ + d];
  const float denom = s + EPS_;
  const float inv_d2 = 1.f / (denom * denom);
  float ge = 0.f, ne = 0.f, k1 = 0.f, t2 = 0.f;
  for (int p = 0; p < P_; ++p) {
    int node = g * P_ + p;
    float v  = nf[(size_t)node * D_ + d];
    float lp = lamb[node * 2 + 0], ln = lamb[node * 2 + 1];
    float nm = lp * v + ln * m;
    float ns = ln * s;
    float u  = unif[(size_t)node * D_ + d];
    ge += v;
    ne += nm + u * ns;
    k1 += 0.5f * ns * ns * inv_d2;
    float t = (nm - m) / denom;
    t2 += t * t;
  }
  out[(size_t)g * D_ + d] = ge;                        // graph_emb
  out[(size_t)G_ * D_ + (size_t)g * D_ + d] = ne;      // noisy_emb
  // --- reduce k1 (sum over features) and t2 (sum over features) ---
  redA[d] = k1; redB[d] = t2;
  __syncthreads();
  for (int st = 64; st > 0; st >>= 1) {
    if (d < st) { redA[d] += redA[d + st]; redB[d] += redB[d + st]; }
    __syncthreads();
  }
  if (d == 0) {
    float kl = redA[0] / (float)(P_ * D_) + redB[0] / (float)D_;
    atomicAdd(&out[(size_t)2 * G_ * D_ + 1], kl / (float)G_);   // all_kl
  }
  __syncthreads();
  // --- preserve: fraction of nodes with assignment[:,0] > 0.5 ---
  redA[d] = (d < P_ && assign[(g * P_ + d) * 2 + 0] > 0.5f) ? 1.f : 0.f;
  __syncthreads();
  for (int st = 64; st > 0; st >>= 1) {
    if (d < st) redA[d] += redA[d + st];
    __syncthreads();
  }
  if (d == 0)
    atomicAdd(&out[(size_t)2 * G_ * D_ + 2],
              redA[0] / (float)P_ / (float)G_);                 // all_preserve
}

// ---------------- per-graph 2x2 adjacency penalty ----------------
__global__ void adj_kernel(const int* __restrict__ esrc,
                           const int* __restrict__ edst,
                           const float* __restrict__ assign,
                           float* __restrict__ out) {
  int g = blockIdx.x * blockDim.x + threadIdx.x;
  if (g >= G_) return;
  float a00 = 0.f, a01 = 0.f, a10 = 0.f, a11 = 0.f;
  int eb = g * EPG_;
  for (int e = 0; e < EPG_; ++e) {
    int sI = esrc[eb + e], dI = edst[eb + e];
    float s0 = assign[sI * 2], s1 = assign[sI * 2 + 1];
    float d0 = assign[dI * 2], d1 = assign[dI * 2 + 1];
    a00 += s0 * d0; a01 += s0 * d1; a10 += s1 * d0; a11 += s1 * d1;
  }
  float r0 = fmaxf(fabsf(a00) + fabsf(a01), 1e-12f);
  float r1 = fmaxf(fabsf(a10) + fabsf(a11), 1e-12f);
  float q0 = a00 / r0 - 1.f, q1 = a11 / r1 - 1.f;
  float pen = 0.5f * (q0 * q0 + q1 * q1);
  atomicAdd(&out[(size_t)2 * G_ * D_ + 0], pen / (float)G_);    // all_pos_penalty
}

// ---------------- launcher ----------------
extern "C" void kernel_launch(void* const* d_in, const int* in_sizes, int n_in,
                              void* d_out, int out_size, void* d_ws, size_t ws_size,
                              hipStream_t stream) {
  const float* x      = (const float*)d_in[0];
  const int*   edge   = (const int*)d_in[1];
  const int*   esrc   = edge;
  const int*   edst   = edge + E_;
  // d_in[2] = batch (implicit from layout; unused)
  const float* gumbel = (const float*)d_in[3];
  const float* unif   = (const float*)d_in[4];
  const float* W1a = (const float*)d_in[5];  const float* b1a = (const float*)d_in[6];
  const float* W1b = (const float*)d_in[7];  const float* b1b = (const float*)d_in[8];
  const float* g1  = (const float*)d_in[9];  const float* be1 = (const float*)d_in[10];
  const float* W2a = (const float*)d_in[11]; const float* b2a = (const float*)d_in[12];
  const float* W2b = (const float*)d_in[13]; const float* b2b = (const float*)d_in[14];
  const float* g2  = (const float*)d_in[15]; const float* be2 = (const float*)d_in[16];
  const float* Wf1 = (const float*)d_in[17]; const float* bf1 = (const float*)d_in[18];
  const float* Wf2 = (const float*)d_in[19]; const float* bf2 = (const float*)d_in[20];
  float* out = (float*)d_out;

  // ---- workspace layout ----
  const size_t NF = (size_t)N_ * D_;
  float* W      = (float*)d_ws;
  float* bufA   = W;                       // z1 -> z2 -> nf
  float* bufB   = W + NF;                  // u1 -> u2
  float* sum1   = W + 2 * NF;              // [512] stats block (zeroed)
  float* sq1    = sum1 + 128;
  float* sum2   = sq1 + 128;
  float* sq2    = sum2 + 128;
  float* scale1 = sq2 + 128;
  float* shift1 = scale1 + 128;
  float* scale2 = shift1 + 128;
  float* shift2 = scale2 + 128;
  float* assignb = shift2 + 128;           // N x 2
  float* lamb    = assignb + 2 * (size_t)N_;
  float* gmean   = lamb + 2 * (size_t)N_;  // G x 128
  float* gstd    = gmean + (size_t)G_ * D_;
  _Float16* Wh   = (_Float16*)(gstd + (size_t)G_ * D_);
  _Float16* W1a_h = Wh;
  _Float16* W1b_h = Wh + 16384;
  _Float16* W2a_h = Wh + 32768;
  _Float16* W2b_h = Wh + 49152;

  // 0) zero accumulators + output scalars
  zero_kernel<<<2, 256, 0, stream>>>(sum1, out + (size_t)2 * G_ * D_);
  // 0b) pack weights into WMMA-fragment-ready f16 layout
  pack_weight_kernel<<<64, 256, 0, stream>>>(W1a, W1a_h);
  pack_weight_kernel<<<64, 256, 0, stream>>>(W1b, W1b_h);
  pack_weight_kernel<<<64, 256, 0, stream>>>(W2a, W2a_h);
  pack_weight_kernel<<<64, 256, 0, stream>>>(W2b, W2b_h);

  // 1) layer-1 aggregation: z1 = x + segsum(x[src]) per graph (LDS atomics)
  gin_aggr_kernel<<<G_, 256, 0, stream>>>(x, esrc, edst, scale1, shift1, 0, bufA);

  // 2) MLP1 (two WMMA GEMMs fused) + BN1 stats
  mlp2_wmma_kernel<<<N_ / (16 * TILES_), 256, 0, stream>>>(
      bufA, W1a_h, b1a, W1b_h, b1b, bufB, sum1, sq1);
  // 3) BN1 params
  bn_params_kernel<<<1, 128, 0, stream>>>(sum1, sq1, g1, be1, scale1, shift1);

  // 4) layer-2 aggregation with BN1+relu fused into the LDS staging
  gin_aggr_kernel<<<G_, 256, 0, stream>>>(bufB, esrc, edst, scale1, shift1, 1, bufA);

  // 5) MLP2 + BN2 stats
  mlp2_wmma_kernel<<<N_ / (16 * TILES_), 256, 0, stream>>>(
      bufA, W2a_h, b2a, W2b_h, b2b, bufB, sum2, sq2);
  // 6) BN2 params, then nf = bn2(u2) (no relu)
  bn_params_kernel<<<1, 128, 0, stream>>>(sum2, sq2, g2, be2, scale2, shift2);
  bn_apply_kernel<<<(int)((NF + 255) / 256), 256, 0, stream>>>(bufB, scale2, shift2, bufA);

  // 7) assignment head
  head_kernel<<<(N_ + 255) / 256, 256, 0, stream>>>(bufA, Wf1, bf1, Wf2, bf2,
                                                    gumbel, assignb, lamb);
  // 8) per-graph mean/std
  gstats_kernel<<<G_, 128, 0, stream>>>(bufA, gmean, gstd);

  // 9) embeddings + KL + preserve
  final_kernel<<<G_, 128, 0, stream>>>(bufA, unif, assignb, lamb, gmean, gstd, out);

  // 10) adjacency diagonal penalty
  adj_kernel<<<(G_ + 255) / 256, 256, 0, stream>>>(esrc, edst, assignb, out);
}
